// TTMultiheadAttention_90228672954977
// MI455X (gfx1250) — compile-verified, matrix operand order checked by reading, not run on hardware
//
#include <hip/hip_runtime.h>

typedef __attribute__((ext_vector_type(16))) _Float16 v16h;
typedef __attribute__((ext_vector_type(8)))  _Float16 v8h;
typedef __attribute__((ext_vector_type(8)))  float    v8f;
typedef __attribute__((ext_vector_type(4)))  float    f4;

#define S_LEN 512
#define D_LEN 256
#define R1 3
#define NFRAG_BR 48              // 3 i * 16 e-tiles
#define NFRAG_CR 8               // 8 k-chunks of 32 over e
#define FRAG_HALVES 512          // 32 lanes * 16 halves
#define TOTAL_HALVES ((NFRAG_BR + NFRAG_CR) * FRAG_HALVES)   // 28672 halves = 57344 B
#define WAVES_PER_BLOCK 4

// K index held by (lane, half h) in the 16-bit A/B WMMA fragment layout:
// lanes 0-15: K = h (h<8) or h+8 (h>=8); lanes 16-31: +8.
__device__ __forceinline__ int frag_k(int lane, int h) {
    return (h < 8 ? h : h + 8) + ((lane >> 4) << 3);
}

// ---------------------------------------------------------------------------
// Prep: swizzle br (scaled by asum[i]/sqrt(32)) and cr into f16 B-fragments.
// br fragment (i, etile): B[k=q (padded to 32), n=e]  = br[i,e,q]*asum[i]*scale
// cr fragment (kchunk c): B[k=e (c*32..+31), n=q]     = cr[q,e]
// ---------------------------------------------------------------------------
__global__ __launch_bounds__(256) void tt_attn_prep(const float* __restrict__ ar,
                                                    const float* __restrict__ br,
                                                    const float* __restrict__ cr,
                                                    _Float16* __restrict__ frag) {
    int idx = blockIdx.x * 256 + threadIdx.x;
    if (idx >= TOTAL_HALVES) return;
    int f    = idx >> 9;          // fragment id
    int slot = idx & 511;
    int lane = slot >> 4;
    int h    = slot & 15;
    int k    = frag_k(lane, h);
    _Float16 val = (_Float16)0.0f;
    if (f < NFRAG_BR) {
        int i = f >> 4, et = f & 15;
        if (k < 16) {             // q dimension is only 16; pad K 16->32 with zeros
            int e = et * 16 + (lane & 15);
            float asum = ar[i] + ar[3 + i] + ar[6 + i];       // ar[(1,3,r1)] flat
            float sc   = asum * 0.17677669529663687f;          // 1/sqrt(head_dim=32)
            val = (_Float16)(br[(i * 256 + e) * 16 + k] * sc);
        }
    } else {
        int c = f - NFRAG_BR;
        int e = c * 32 + k;
        val = (_Float16)cr[(lane & 15) * 256 + e];             // cr[(16,256,1)] flat
    }
    frag[idx] = val;
}

// ---------------------------------------------------------------------------
// Main: one wave handles one (s, d-tile) task.
//   stage1: scores[d16, e256] = A(x) x B(br_i)   via 16x v_wmma per i
//   softmax (no max-sub, single pass): p = exp(score), Z = rowsum(p)
//   stage2: out[d16, q16] += (p/Z) x B(cr)       via  8x v_wmma per i
// ---------------------------------------------------------------------------
__global__ __launch_bounds__(128) void tt_attn_main(const float* __restrict__ x,
                                                    const _Float16* __restrict__ frag,
                                                    float* __restrict__ out) {
    __shared__ _Float16 fragS[TOTAL_HALVES];                       // 57344 B
    __shared__ __align__(16) float scratchS[WAVES_PER_BLOCK][256]; //  4096 B

    // Async stage of all B-fragments into LDS: GLOBAL_LOAD_ASYNC_TO_LDS_B128
    // (VGPR-bypassing DMA path, tracked with ASYNCcnt). Low 32 bits of a
    // generic pointer to LDS are the wave-relative LDS address (ISA 10.2).
    {
        unsigned    lbase = (unsigned)(uintptr_t)fragS;
        const char* gbase = (const char*)frag;
        for (int j = threadIdx.x; j < TOTAL_HALVES / 8; j += 128) {
            unsigned           laddr = lbase + (unsigned)(j * 16);
            unsigned long long gaddr = (unsigned long long)(uintptr_t)(gbase + j * 16);
            asm volatile("global_load_async_to_lds_b128 %0, %1, off"
                         :: "v"(laddr), "v"(gaddr) : "memory");
        }
        asm volatile("s_wait_asynccnt 0" ::: "memory");
    }
    __syncthreads();

    const int lane = threadIdx.x & 31;
    const int wave = threadIdx.x >> 5;
    const int task = blockIdx.x * WAVES_PER_BLOCK + wave;   // 8192 tasks
    const int s    = task >> 4;
    const int dt   = task & 15;
    const int hi   = lane >> 4;       // which 16-lane half
    const int lm   = lane & 15;

    _Float16* pS = (_Float16*)scratchS[wave];   // 512 halves: P transpose tile
    float*    tS = scratchS[wave];              // 256 floats: output transpose tile

    // A fragment: A[m=d_local, k=q], K padded 16->32 with zeros.
    // lane lm holds row m=lm; per-q loads are 64B-contiguous across lanes.
    v16h aX;
    const float* xp = x + (size_t)s * D_LEN + dt * 16 + lm;
#pragma unroll
    for (int h = 0; h < 8; ++h) {
        int q = h + hi * 8;                                  // == frag_k(lane,h), h<8
        aX[h]     = (_Float16)xp[(size_t)q * (S_LEN * D_LEN)];
        aX[h + 8] = (_Float16)0.0f;
    }

    v8f outAcc = {};

    for (int i = 0; i < R1; ++i) {
        v8f   c2 = {};
        float rs[8];
#pragma unroll
        for (int v = 0; v < 8; ++v) rs[v] = 0.0f;

        for (int t = 0; t < 8; ++t) {    // pairs of e-tiles == one K=32 chunk of e
            const v16h b0 = *(const v16h*)&fragS[(i * 16 + 2 * t    ) * FRAG_HALVES + lane * 16];
            const v16h b1 = *(const v16h*)&fragS[(i * 16 + 2 * t + 1) * FRAG_HALVES + lane * 16];
            v8f z = {};
            v8f c0 = __builtin_amdgcn_wmma_f32_16x16x32_f16(false, aX, false, b0,
                                                            (short)0, z, false, false);
            v8f c1 = __builtin_amdgcn_wmma_f32_16x16x32_f16(false, aX, false, b1,
                                                            (short)0, z, false, false);

            asm volatile("" ::: "memory");
            // exp + rowsum accumulate + write P tile [m][e_local] in f16
#pragma unroll
            for (int v = 0; v < 8; ++v) {
                float p0 = __expf(c0[v]);
                float p1 = __expf(c1[v]);
                rs[v] += p0 + p1;
                int m = v + hi * 8;
                pS[m * 32 + lm]      = (_Float16)p0;
                pS[m * 32 + 16 + lm] = (_Float16)p1;
            }
            asm volatile("s_wait_dscnt 0" ::: "memory");

            // Re-read P in A-fragment layout (two 16B DS reads per lane).
            v8h lo  = *(const v8h*)&pS[lm * 32 + hi * 8];
            v8h hi8 = *(const v8h*)&pS[lm * 32 + 16 + hi * 8];
            v16h pf;
#pragma unroll
            for (int j = 0; j < 8; ++j) { pf[j] = lo[j]; pf[j + 8] = hi8[j]; }

            const v16h cf = *(const v16h*)&fragS[(NFRAG_BR + t) * FRAG_HALVES + lane * 16];
            c2 = __builtin_amdgcn_wmma_f32_16x16x32_f16(false, pf, false, cf,
                                                        (short)0, c2, false, false);
            asm volatile("" ::: "memory");   // keep pS reads ahead of next overwrite
        }

        // Row-sum reduction across the 16-lane halves (masks 1,2,4,8 stay in-half),
        // then normalize this i's contribution and accumulate.
#pragma unroll
        for (int v = 0; v < 8; ++v) {
            float r = rs[v];
            r += __shfl_xor(r, 1, 32);
            r += __shfl_xor(r, 2, 32);
            r += __shfl_xor(r, 4, 32);
            r += __shfl_xor(r, 8, 32);
            outAcc[v] += c2[v] / r;
        }
    }

    // Transpose C layout [m across vgpr, q across lanes] -> [q][m] via LDS,
    // then fully coalesced 32B stores per lane (2 lanes cover each q row).
    asm volatile("" ::: "memory");
#pragma unroll
    for (int v = 0; v < 8; ++v)
        tS[lm * 16 + hi * 8 + v] = outAcc[v];
    asm volatile("s_wait_dscnt 0" ::: "memory");

    f4 r0 = *(const f4*)&tS[lane * 8];
    f4 r1 = *(const f4*)&tS[lane * 8 + 4];
    const int q = lane >> 1;
    float* gp = out + (size_t)q * (S_LEN * D_LEN) + (size_t)s * D_LEN
                    + dt * 16 + (lane & 1) * 8;
    *(f4*)gp       = r0;
    *((f4*)gp + 1) = r1;
}

extern "C" void kernel_launch(void* const* d_in, const int* in_sizes, int n_in,
                              void* d_out, int out_size, void* d_ws, size_t ws_size,
                              hipStream_t stream) {
    const float* x  = (const float*)d_in[0];
    const float* ar = (const float*)d_in[1];
    const float* br = (const float*)d_in[2];
    const float* cr = (const float*)d_in[3];
    _Float16* frag = (_Float16*)d_ws;       // 57344 B of scratch used
    float* out = (float*)d_out;

    tt_attn_prep<<<(TOTAL_HALVES + 255) / 256, 256, 0, stream>>>(ar, br, cr, frag);

    const int tasks = S_LEN * (D_LEN / 16);                  // 8192
    tt_attn_main<<<tasks / WAVES_PER_BLOCK, 128, 0, stream>>>(x, frag, out);
}
